// Attention_29283087024612
// MI455X (gfx1250) — matrix operand order repaired and checked by reading, not compiled, not run on hardware
//
#include <hip/hip_runtime.h>
#include <stdint.h>

typedef __attribute__((ext_vector_type(16))) __bf16 v16bf;
typedef __attribute__((ext_vector_type(8)))  float  v8f;

union Frag16 { v16bf v; uint4 q[2]; };

__device__ inline __bf16 f2bf(float x) {
    union { float f; uint32_t u; } v; v.f = x;
    uint32_t r = (v.u + 0x7FFFu + ((v.u >> 16) & 1u)) >> 16;  // round-nearest-even
    union { unsigned short s; __bf16 b; } o; o.s = (unsigned short)r;
    return o.b;
}
__device__ inline float bf2f(__bf16 b) {
    union { unsigned short s; __bf16 b; } t; t.b = b;
    union { uint32_t u; float f; } v; v.u = ((uint32_t)t.s) << 16;
    return v.f;
}
__device__ inline void bfstore2(__bf16* p, float a, float b) {  // p 4B-aligned
    union { unsigned short s; __bf16 b; } ua, ub;
    ua.b = f2bf(a); ub.b = f2bf(b);
    *(uint32_t*)p = ((uint32_t)ub.s << 16) | ua.s;
}

// ---- CDNA5 async global->LDS copy (ASYNCcnt-tracked) ----------------------
__device__ inline void async_b128(const void* g, const void* l) {
    unsigned lds = (unsigned)(size_t)l;                  // low 32 bits = LDS offset
    unsigned long long ga = (unsigned long long)(size_t)g;
    asm volatile("global_load_async_to_lds_b128 %0, %1, off"
                 :: "v"(lds), "v"(ga) : "memory");
}
__device__ inline void wait_async() {
    asm volatile("s_wait_asynccnt 0x0" ::: "memory");
}

// ---------------------------------------------------------------------------
// Elementwise prep kernels (run once per launch; bandwidth-trivial).
// ---------------------------------------------------------------------------
__global__ __launch_bounds__(256) void cvt_bf16(const float* __restrict__ in,
                                                __bf16* __restrict__ out, int n) {
    int tid = blockIdx.x * 256 + threadIdx.x;
    if (tid < n) out[tid] = f2bf(in[tid]);
}

// w [2048, N] f32  ->  wt [N, 2048] bf16 (row-major, K contiguous)
__global__ __launch_bounds__(256) void transpose_w(const float* __restrict__ w,
                                                   __bf16* __restrict__ wt, int N) {
    int tid = blockIdx.x * 256 + threadIdx.x;            // over N*2048
    if (tid >= N * 2048) return;
    int k = tid & 2047;
    int n = tid >> 11;
    wt[tid] = f2bf(w[(size_t)k * N + n]);
}

// RoPE in place on bf16 q [B,S,16,128] (with softmax scale folded) and k [B,S,4,128].
__global__ __launch_bounds__(256) void rope_bf16(
    __bf16* __restrict__ q, __bf16* __restrict__ k,
    const float* __restrict__ fc, const float* __restrict__ fs)
{
    const int NQ = 2 * 2048 * 16 * 64;
    const int NK = 2 * 2048 * 4 * 64;
    int tid = blockIdx.x * 256 + threadIdx.x;
    if (tid < NQ) {
        int d2 = tid & 63, h = (tid >> 6) & 15, s = (tid >> 10) & 2047, b = tid >> 21;
        float c = fc[s * 64 + d2], sn = fs[s * 64 + d2];
        size_t off = ((size_t)(b * 2048 + s)) * 2048 + h * 128 + 2 * d2;
        float te = bf2f(q[off]), to = bf2f(q[off + 1]);
        const float sc = 0.08838834764831845f;           // 1/sqrt(128)
        bfstore2(q + off, (te * c - to * sn) * sc, (te * sn + to * c) * sc);
    } else if (tid < NQ + NK) {
        int t = tid - NQ;
        int d2 = t & 63, h = (t >> 6) & 3, s = (t >> 8) & 2047, b = t >> 19;
        float c = fc[s * 64 + d2], sn = fs[s * 64 + d2];
        size_t off = ((size_t)(b * 2048 + s)) * 512 + h * 128 + 2 * d2;
        float te = bf2f(k[off]), to = bf2f(k[off + 1]);
        bfstore2(k + off, te * c - to * sn, te * sn + to * c);
    }
}

// v [B,S,4,128] bf16 -> vt [(b*4+kvh)*128+d][S] bf16  (keys contiguous per d-row)
__global__ __launch_bounds__(256) void transpose_v(const __bf16* __restrict__ v,
                                                   __bf16* __restrict__ vt) {
    int tid = blockIdx.x * 256 + threadIdx.x;            // over 2*4*128*2048
    int s = tid & 2047, d = (tid >> 11) & 127, kvh = (tid >> 18) & 3, b = tid >> 20;
    vt[tid] = v[((size_t)(b * 2048 + s)) * 512 + kvh * 128 + d];
}

// ---------------------------------------------------------------------------
// GEMM: C[M,N] = A[M,K] (bf16) @ Wt[N,K]^T (bf16), f32 accum, async-staged,
// double-buffered LDS. 128 threads / 4 waves, block tile 64x64, K-step 32.
// ---------------------------------------------------------------------------
template <bool OUTBF>
__global__ __launch_bounds__(128) void gemm_async(
    const __bf16* __restrict__ A, const __bf16* __restrict__ Wt,
    float* __restrict__ Cf, __bf16* __restrict__ Cb, int N, int K)
{
    __shared__ __bf16 Alds[2][64 * 32];
    __shared__ __bf16 Wlds[2][64 * 32];

    const int tid   = threadIdx.x;
    const int lane  = tid & 31;
    const int wave  = tid >> 5;
    const int tileM = blockIdx.y * 64;
    const int tileN = blockIdx.x * 64;
    const int waveM = (wave >> 1) * 32;
    const int waveN = (wave & 1) * 32;

    auto issue = [&](int k0, int buf) {
        #pragma unroll
        for (int i = 0; i < 2; ++i) {
            int c = tid + i * 128;                       // 256 x 16B chunks each
            int row = c >> 2, seg = (c & 3) * 8;
            async_b128(A  + (size_t)(tileM + row) * K + k0 + seg,
                       &Alds[buf][row * 32 + seg]);
            async_b128(Wt + (size_t)(tileN + row) * K + k0 + seg,
                       &Wlds[buf][row * 32 + seg]);
        }
    };

    v8f c00 = {}; v8f c01 = {}; v8f c10 = {}; v8f c11 = {};
    issue(0, 0);
    const int nK = K >> 5;
    for (int it = 0; it < nK; ++it) {
        wait_async();
        __syncthreads();
        if (it + 1 < nK) issue((it + 1) << 5, (it + 1) & 1);

        const __bf16* Ald = Alds[it & 1];
        const __bf16* Wld = Wlds[it & 1];
        Frag16 a0, a1, b0, b1;
        {
            int row = waveM + (lane & 15);
            int c0  = (lane >> 4) * 8;
            const __bf16* p = Ald + row * 32;
            a0.q[0] = *(const uint4*)(p + c0);
            a0.q[1] = *(const uint4*)(p + c0 + 16);
            p += 16 * 32;
            a1.q[0] = *(const uint4*)(p + c0);
            a1.q[1] = *(const uint4*)(p + c0 + 16);
        }
        {
            int n  = waveN + (lane & 15);
            int kb = (lane >> 4) * 8;
            const __bf16* p = Wld + n * 32 + kb;
            b0.q[0] = *(const uint4*)(p);
            b0.q[1] = *(const uint4*)(p + 16);
            p += 16 * 32;
            b1.q[0] = *(const uint4*)(p);
            b1.q[1] = *(const uint4*)(p + 16);
        }
        c00 = __builtin_amdgcn_wmma_f32_16x16x32_bf16(false, a0.v, false, b0.v, (short)0, c00, false, false);
        c01 = __builtin_amdgcn_wmma_f32_16x16x32_bf16(false, a0.v, false, b1.v, (short)0, c01, false, false);
        c10 = __builtin_amdgcn_wmma_f32_16x16x32_bf16(false, a1.v, false, b0.v, (short)0, c10, false, false);
        c11 = __builtin_amdgcn_wmma_f32_16x16x32_bf16(false, a1.v, false, b1.v, (short)0, c11, false, false);
    }

    int rbase = tileM + waveM + ((lane >> 4) << 3);
    int cbase = tileN + waveN + (lane & 15);
    #pragma unroll
    for (int r = 0; r < 8; ++r) {
        if constexpr (OUTBF) {
            Cb[(size_t)(rbase + r)      * N + cbase]      = f2bf(c00[r]);
            Cb[(size_t)(rbase + r)      * N + cbase + 16] = f2bf(c01[r]);
            Cb[(size_t)(rbase + 16 + r) * N + cbase]      = f2bf(c10[r]);
            Cb[(size_t)(rbase + 16 + r) * N + cbase + 16] = f2bf(c11[r]);
        } else {
            Cf[(size_t)(rbase + r)      * N + cbase]      = c00[r];
            Cf[(size_t)(rbase + r)      * N + cbase + 16] = c01[r];
            Cf[(size_t)(rbase + 16 + r) * N + cbase]      = c10[r];
            Cf[(size_t)(rbase + 16 + r) * N + cbase + 16] = c11[r];
        }
    }
}

// ---------------------------------------------------------------------------
// Flash attention, all-bf16 operands (pre-scaled/roped), async-staged tiles.
// One block per (64 q rows, head, batch); 4 waves, 16 q rows each.
// ---------------------------------------------------------------------------
__global__ __launch_bounds__(128) void attn_kernel(
    const __bf16* __restrict__ Qb, const __bf16* __restrict__ Kb,
    const __bf16* __restrict__ Vt, __bf16* __restrict__ Ob)
{
    __shared__ __bf16 Qs[64 * 128];       // [qrow][d]
    __shared__ __bf16 Ks[32 * 128];       // [key][d]
    __shared__ __bf16 Vs[128 * 32];       // [d][key]
    __shared__ __bf16 Ps[4 * 16 * 32];    // per-wave P tile [qrow][key]

    const int tid  = threadIdx.x;
    const int lane = tid & 31;
    const int wave = tid >> 5;
    const int qblk = blockIdx.x;
    const int h    = blockIdx.y;
    const int b    = blockIdx.z;
    const int kvh  = h & 3;               // jnp.tile => kv head = h % 4
    const int S    = 2048;

    // Stage Q tile 64x128 via async copies (1024 x 16B chunks).
    #pragma unroll
    for (int i = 0; i < 8; ++i) {
        int c = tid + i * 128;
        int row = c >> 4, seg = (c & 15) * 8;
        async_b128(Qb + ((size_t)(b * S + qblk * 64 + row)) * 2048 + h * 128 + seg,
                   &Qs[row * 128 + seg]);
    }
    wait_async();
    __syncthreads();

    Frag16 qf[4];
    {
        int row = wave * 16 + (lane & 15);
        int c0  = (lane >> 4) * 8;
        const __bf16* p = Qs + row * 128;
        #pragma unroll
        for (int c = 0; c < 4; ++c) {
            qf[c].q[0] = *(const uint4*)(p + c * 32 + c0);
            qf[c].q[1] = *(const uint4*)(p + c * 32 + c0 + 16);
        }
    }

    v8f acc[8] = {};
    float mrow[8], lrow[8];
    #pragma unroll
    for (int r = 0; r < 8; ++r) { mrow[r] = -1e30f; lrow[r] = 0.0f; }

    const int jmax = qblk * 2 + 1;        // causal tile bound
    for (int j = 0; j <= jmax; ++j) {
        __syncthreads();
        // Async-stage K (32x128) and Vt rows (128 x 32 keys): 4+4 chunks/thread.
        #pragma unroll
        for (int i = 0; i < 4; ++i) {
            int c = tid + i * 128;
            int key = c >> 4, seg = (c & 15) * 8;
            const __bf16* gk = Kb + ((size_t)(b * S + j * 32 + key)) * 512 + kvh * 128 + seg;
            async_b128(gk, &Ks[key * 128 + seg]);
            if (i == 0 && j < jmax) __builtin_prefetch(gk + 32 * 512, 0, 1);
            int d = c >> 2, seg2 = (c & 3) * 8;
            async_b128(Vt + ((size_t)((b * 4 + kvh) * 128 + d)) * 2048 + j * 32 + seg2,
                       &Vs[d * 32 + seg2]);
        }
        wait_async();
        __syncthreads();

        // scores: 8 WMMAs.
        v8f s0 = {}; v8f s1 = {};
        {
            int n  = lane & 15;
            int kb = (lane >> 4) * 8;
            #pragma unroll
            for (int c = 0; c < 4; ++c) {
                Frag16 kf0, kf1;
                const __bf16* p0 = Ks + n * 128 + c * 32 + kb;
                kf0.q[0] = *(const uint4*)(p0);
                kf0.q[1] = *(const uint4*)(p0 + 16);
                const __bf16* p1 = Ks + (16 + n) * 128 + c * 32 + kb;
                kf1.q[0] = *(const uint4*)(p1);
                kf1.q[1] = *(const uint4*)(p1 + 16);
                s0 = __builtin_amdgcn_wmma_f32_16x16x32_bf16(false, qf[c].v, false, kf0.v, (short)0, s0, false, false);
                s1 = __builtin_amdgcn_wmma_f32_16x16x32_bf16(false, qf[c].v, false, kf1.v, (short)0, s1, false, false);
            }
        }

        // Causal mask + online softmax.
        int col    = lane & 15;
        int rowoff = qblk * 64 + wave * 16 + ((lane >> 4) << 3);
        int key0   = j * 32 + col;
        int key1   = key0 + 16;
        #pragma unroll
        for (int r = 0; r < 8; ++r) {
            int grow = rowoff + r;
            float v0 = s0[r] + ((key0 > grow) ? -1e9f : 0.0f);
            float v1 = s1[r] + ((key1 > grow) ? -1e9f : 0.0f);
            float t = fmaxf(v0, v1);
            t = fmaxf(t, __shfl_xor(t, 1, 32));
            t = fmaxf(t, __shfl_xor(t, 2, 32));
            t = fmaxf(t, __shfl_xor(t, 4, 32));
            t = fmaxf(t, __shfl_xor(t, 8, 32));
            float mnew = fmaxf(mrow[r], t);
            float corr = __expf(mrow[r] - mnew);
            float p0 = __expf(v0 - mnew);
            float p1 = __expf(v1 - mnew);
            float ps = p0 + p1;
            ps += __shfl_xor(ps, 1, 32);
            ps += __shfl_xor(ps, 2, 32);
            ps += __shfl_xor(ps, 4, 32);
            ps += __shfl_xor(ps, 8, 32);
            lrow[r] = lrow[r] * corr + ps;
            mrow[r] = mnew;
            #pragma unroll
            for (int n = 0; n < 8; ++n) acc[n][r] *= corr;
            int Mr = ((lane >> 4) << 3) + r;
            Ps[wave * 512 + Mr * 32 + col]      = f2bf(p0);
            Ps[wave * 512 + Mr * 32 + col + 16] = f2bf(p1);
        }

        // O += P @ V : 8 WMMAs.
        Frag16 pf;
        {
            int row = lane & 15;
            int c0  = (lane >> 4) * 8;
            const __bf16* p = Ps + wave * 512 + row * 32;
            pf.q[0] = *(const uint4*)(p + c0);
            pf.q[1] = *(const uint4*)(p + c0 + 16);
        }
        {
            int n  = lane & 15;
            int kb = (lane >> 4) * 8;
            #pragma unroll
            for (int nt = 0; nt < 8; ++nt) {
                Frag16 vf;
                const __bf16* p = Vs + (nt * 16 + n) * 32 + kb;
                vf.q[0] = *(const uint4*)(p);
                vf.q[1] = *(const uint4*)(p + 16);
                acc[nt] = __builtin_amdgcn_wmma_f32_16x16x32_bf16(false, pf.v, false, vf.v, (short)0, acc[nt], false, false);
            }
        }
    }

    // Finalize: bf16 attention output [B*S, 2048] for the final GEMM's A operand.
    {
        int col    = lane & 15;
        int rowoff = qblk * 64 + wave * 16 + ((lane >> 4) << 3);
        #pragma unroll
        for (int r = 0; r < 8; ++r) {
            float inv = 1.0f / lrow[r];
            size_t base = ((size_t)(b * S + rowoff + r)) * 2048 + h * 128;
            #pragma unroll
            for (int nt = 0; nt < 8; ++nt)
                Ob[base + nt * 16 + col] = f2bf(acc[nt][r] * inv);
        }
    }
}

// ---------------------------------------------------------------------------
extern "C" void kernel_launch(void* const* d_in, const int* in_sizes, int n_in,
                              void* d_out, int out_size, void* d_ws, size_t ws_size,
                              hipStream_t stream)
{
    const float* x  = (const float*)d_in[0];
    const float* wq = (const float*)d_in[1];
    const float* wk = (const float*)d_in[2];
    const float* wv = (const float*)d_in[3];
    const float* wo = (const float*)d_in[4];
    const float* fc = (const float*)d_in[5];
    const float* fs = (const float*)d_in[6];
    float* out = (float*)d_out;

    const int M = 2 * 2048;                               // B*S
    __bf16* xbf = (__bf16*)d_ws;                          // [M, 2048]
    __bf16* wqT = xbf + (size_t)M * 2048;                 // [2048, 2048]
    __bf16* wkT = wqT + (size_t)2048 * 2048;              // [512, 2048]
    __bf16* wvT = wkT + (size_t)512 * 2048;               // [512, 2048]
    __bf16* woT = wvT + (size_t)512 * 2048;               // [2048, 2048]
    __bf16* qbf = woT + (size_t)2048 * 2048;              // [M, 2048]
    __bf16* kbf = qbf + (size_t)M * 2048;                 // [M, 512]
    __bf16* vbf = kbf + (size_t)M * 512;                  // [M, 512]
    __bf16* vT  = vbf + (size_t)M * 512;                  // [2*4*128, 2048]
    __bf16* ab  = vT  + (size_t)M * 512;                  // [M, 2048]

    // 1) Pack inputs to bf16 (weights transposed to [N,K]).
    cvt_bf16<<<(M * 2048) / 256, 256, 0, stream>>>(x, xbf, M * 2048);
    transpose_w<<<(2048 * 2048) / 256, 256, 0, stream>>>(wq, wqT, 2048);
    transpose_w<<<(512  * 2048) / 256, 256, 0, stream>>>(wk, wkT, 512);
    transpose_w<<<(512  * 2048) / 256, 256, 0, stream>>>(wv, wvT, 512);
    transpose_w<<<(2048 * 2048) / 256, 256, 0, stream>>>(wo, woT, 2048);

    // 2) Projections (bf16 out).
    dim3 blk(128);
    gemm_async<true><<<dim3(2048 / 64, M / 64), blk, 0, stream>>>(xbf, wqT, nullptr, qbf, 2048, 2048);
    gemm_async<true><<<dim3(512 / 64,  M / 64), blk, 0, stream>>>(xbf, wkT, nullptr, kbf, 512,  2048);
    gemm_async<true><<<dim3(512 / 64,  M / 64), blk, 0, stream>>>(xbf, wvT, nullptr, vbf, 512,  2048);

    // 3) RoPE (+softmax scale) in place; V transposed for P@V staging.
    const int nrope = 2 * 2048 * 16 * 64 + 2 * 2048 * 4 * 64;
    rope_bf16<<<(nrope + 255) / 256, 256, 0, stream>>>(qbf, kbf, fc, fs);
    transpose_v<<<(M * 512) / 256, 256, 0, stream>>>(vbf, vT);

    // 4) Attention (bf16 out).
    attn_kernel<<<dim3(32, 16, 2), blk, 0, stream>>>(qbf, kbf, vT, ab);

    // 5) Output projection (f32 out).
    gemm_async<false><<<dim3(2048 / 64, M / 64), blk, 0, stream>>>(ab, woT, out, nullptr, 2048, 2048);
}
